// SelfAttn_35699768164707
// MI455X (gfx1250) — matrix operand order repaired and checked by reading, not compiled, not run on hardware
//
#include <hip/hip_runtime.h>
#include <hip/hip_bf16.h>
#include <cstdint>

typedef __attribute__((ext_vector_type(16))) __bf16 v16bf;
typedef __attribute__((ext_vector_type(8)))  __bf16 v8bf;
typedef __attribute__((ext_vector_type(8)))  float  v8f;
typedef __attribute__((ext_vector_type(4)))  float  v4f;
typedef __attribute__((ext_vector_type(4)))  unsigned int v4ui;
typedef __attribute__((ext_vector_type(8)))  int    v8i;
typedef __attribute__((ext_vector_type(4)))  int    v4i;

#define WINDOW 2048
#define EMBED  1024
#define BATCH  8

// ---------------------------------------------------------------------------
// Fragment loaders. CDNA5 WMMA bf16 16x16x32 per-lane layouts (ISA 7.12.2):
//  A (16xK=32): lane&15 = row M; lanes 0-15 hold K {0..7,16..23},
//               lanes 16-31 hold K {8..15,24..31}  -> two contiguous 8-runs.
//  B (K=32x16): lane&15 = col N; lanes 0-15 hold K {0..7,16..23}, lanes 16-31 K {8..15,24..31}
//               -> two contiguous 8-runs, same as A.
// ---------------------------------------------------------------------------

__device__ inline v16bf loadA_bf16(const __bf16* src, int ld,
                                   int rb, int kb, int lane) {
    const __bf16* p = src + (size_t)(rb + (lane & 15)) * ld + kb + ((lane >> 4) << 3);
    v8bf lo = *(const v8bf*)(p);        // K = ch .. ch+7
    v8bf hi = *(const v8bf*)(p + 16);   // K = ch+16 .. ch+23
    v16bf r;
#pragma unroll
    for (int i = 0; i < 8; ++i) { r[i] = lo[i]; r[8 + i] = hi[i]; }
    return r;
}

__device__ inline v16bf loadB_bf16(const __bf16* src, int ld,
                                   int cb, int kb, int lane) {
    // src layout: element (k, n) at src[(cb+n)*ld + kb + k]
    const __bf16* p = src + (size_t)(cb + (lane & 15)) * ld + kb + ((lane >> 4) << 3);
    v8bf lo = *(const v8bf*)(p);
    v8bf hi = *(const v8bf*)(p + 16);
    v16bf r;
#pragma unroll
    for (int i = 0; i < 8; ++i) { r[i] = lo[i]; r[8 + i] = hi[i]; }
    return r;
}

__device__ inline v16bf loadA_f32(const float* __restrict__ src, int ld,
                                  int rb, int kb, int lane) {
    const float* p = src + (size_t)(rb + (lane & 15)) * ld + kb + ((lane >> 4) << 3);
    v4f a0 = *(const v4f*)(p);
    v4f a1 = *(const v4f*)(p + 4);
    v4f a2 = *(const v4f*)(p + 16);
    v4f a3 = *(const v4f*)(p + 20);
    v16bf r;
#pragma unroll
    for (int i = 0; i < 4; ++i) {
        r[i]      = (__bf16)a0[i];
        r[4 + i]  = (__bf16)a1[i];
        r[8 + i]  = (__bf16)a2[i];
        r[12 + i] = (__bf16)a3[i];
    }
    return r;
}

__device__ inline v16bf loadB_f32(const float* __restrict__ src, int ld,
                                  int cb, int kb, int lane) {
    const float* p = src + (size_t)(cb + (lane & 15)) * ld + kb + ((lane >> 4) << 3);
    v4f a0 = *(const v4f*)(p);
    v4f a1 = *(const v4f*)(p + 4);
    v4f a2 = *(const v4f*)(p + 16);
    v4f a3 = *(const v4f*)(p + 20);
    v16bf r;
#pragma unroll
    for (int i = 0; i < 4; ++i) {
        r[i]      = (__bf16)a0[i];
        r[4 + i]  = (__bf16)a1[i];
        r[8 + i]  = (__bf16)a2[i];
        r[12 + i] = (__bf16)a3[i];
    }
    return r;
}

#define WMMA_BF16(a, b, c) \
    __builtin_amdgcn_wmma_f32_16x16x32_bf16(false, (a), false, (b), (short)0, (c), false, false)

// ---------------------------------------------------------------------------
// TDM helper: 2-D tensor_load_to_lds.  Tile = tile0 x tile1 elements of 2B
// data; tensor line length tdim0 (elements), line stride stride0 (elements).
// LDS receives tile1 consecutive rows of tile0 elements.
// D# layout per CDNA5 ISA.
// ---------------------------------------------------------------------------
__device__ inline void tdm_load_2d_bf16(unsigned lds_off, const void* gptr,
                                        unsigned tdim0, unsigned tdim1,
                                        unsigned tile0, unsigned tile1,
                                        unsigned stride0) {
    uint64_t ga = (uint64_t)(uintptr_t)gptr;
    v4ui g0;
    g0[0] = 1u;                                   // count=1 (valid descriptor)
    g0[1] = lds_off;                              // lds_addr (bytes)
    g0[2] = (unsigned)(ga & 0xffffffffu);         // global_addr[31:0]
    g0[3] = (unsigned)((ga >> 32) & 0x1ffffffu)   // global_addr[56:32]
          | (2u << 30);                           // type=2 ("image")
    v8i g1;
    g1[0] = (int)(1u << 16);                      // data_size=1 -> 2 bytes
    g1[1] = (int)(tdim0 << 16);                   // tensor_dim0[15:0] @ bits 63:48
    g1[2] = (int)((tdim0 >> 16) | (tdim1 << 16)); // dim0 hi | dim1 lo
    g1[3] = (int)((tdim1 >> 16) | (tile0 << 16)); // dim1 hi | tile_dim0
    g1[4] = (int)(tile1 & 0xffffu);               // tile_dim1, tile_dim2=0
    g1[5] = (int)stride0;                         // tensor_dim0_stride lo
    g1[6] = 0;                                    // stride hi | dim1_stride lo
    g1[7] = 0;
    v4i z4 = {0, 0, 0, 0};
#if __clang_major__ >= 23
    v8i z8 = {0, 0, 0, 0, 0, 0, 0, 0};
    __builtin_amdgcn_tensor_load_to_lds(g0, g1, z4, z4, z8, 0);
#else
    __builtin_amdgcn_tensor_load_to_lds(g0, g1, z4, z4, 0);
#endif
}

// ---------------------------------------------------------------------------
// Kernel 1: QKV projections.  Y = X @ W^T + b  -> bf16.
// z=0: Q (scaled by 1/sqrt(E)), row-major.  z=1: K row-major.
// z=2: V stored transposed Vt[b][e][s].
// ---------------------------------------------------------------------------
__global__ __launch_bounds__(256) void qkv_kernel(
    const float* __restrict__ X,
    const float* __restrict__ Wq, const float* __restrict__ bq,
    const float* __restrict__ Wk, const float* __restrict__ bk,
    const float* __restrict__ Wv, const float* __restrict__ bv,
    __bf16* __restrict__ Qb, __bf16* __restrict__ Kb, __bf16* __restrict__ Vt)
{
    const int z = blockIdx.z;
    const float* W    = (z == 0) ? Wq : (z == 1) ? Wk : Wv;
    const float* bias = (z == 0) ? bq : (z == 1) ? bk : bv;
    const float outScale = (z == 0) ? 0.03125f : 1.0f;  // 1/sqrt(1024)

    const int lane = threadIdx.x & 31;
    const int wave = threadIdx.x >> 5;
    const int mrow = blockIdx.x * 32 + (wave & 1) * 16;
    const int ncol = blockIdx.y * 64 + (wave >> 1) * 16;

    const float bval = bias[ncol + (lane & 15)];
    v8f acc;
#pragma unroll
    for (int i = 0; i < 8; ++i) acc[i] = bval;

#pragma unroll 2
    for (int kk = 0; kk < EMBED; kk += 32) {
        v16bf a = loadA_f32(X, EMBED, mrow, kk, lane);
        v16bf b = loadB_f32(W, EMBED, ncol, kk, lane);
        acc = WMMA_BF16(a, b, acc);
    }

    const int n  = ncol + (lane & 15);
    const int mb = mrow + ((lane >> 4) ? 8 : 0);
    if (z == 2) {
#pragma unroll
        for (int r = 0; r < 8; ++r) {
            int m = mb + r;
            int bidx = m >> 11, s = m & (WINDOW - 1);
            Vt[((size_t)bidx * EMBED + n) * WINDOW + s] = (__bf16)acc[r];
        }
    } else {
        __bf16* O = (z == 0) ? Qb : Kb;
#pragma unroll
        for (int r = 0; r < 8; ++r)
            O[(size_t)(mb + r) * EMBED + n] = (__bf16)(acc[r] * outScale);
    }
}

// ---------------------------------------------------------------------------
// Kernel 2: per-column (key index j) softmax stats over the query axis.
// One wave owns 16 columns of one batch; walks the causal i-strip with WMMA
// QK^T tiles, keeping online (max, sumexp) per column.
// ---------------------------------------------------------------------------
__global__ __launch_bounds__(256) void stats_kernel(
    const __bf16* __restrict__ Qb, const __bf16* __restrict__ Kb,
    float* __restrict__ colmax, float* __restrict__ colsum)
{
    const int lane = threadIdx.x & 31;
    const int gw = blockIdx.x * 8 + (threadIdx.x >> 5);  // 0..1023
    const int b  = gw >> 7;
    const int jt = gw & 127;
    const __bf16* Q = Qb + (size_t)b * WINDOW * EMBED;
    const __bf16* K = Kb + (size_t)b * WINDOW * EMBED;

    const int jcol = jt * 16 + (lane & 15);
    float m = -__builtin_inff();
    float zsum = 0.0f;

    for (int it = jt; it < WINDOW / 16; ++it) {
        v8f acc;
#pragma unroll
        for (int i = 0; i < 8; ++i) acc[i] = 0.0f;
#pragma unroll 2
        for (int kk = 0; kk < EMBED; kk += 32) {
            v16bf a = loadA_bf16(Q, EMBED, it * 16, kk, lane);
            v16bf bb = loadB_bf16(K, EMBED, jt * 16, kk, lane);
            acc = WMMA_BF16(a, bb, acc);
        }
        const int ibase = it * 16 + ((lane >> 4) ? 8 : 0);
        float vm = -__builtin_inff();
#pragma unroll
        for (int r = 0; r < 8; ++r)
            if (ibase + r >= jcol) vm = fmaxf(vm, acc[r]);
        const float newm = fmaxf(m, vm);
        if (newm != -__builtin_inff()) {
            float zl = zsum * __expf(m - newm);
#pragma unroll
            for (int r = 0; r < 8; ++r)
                if (ibase + r >= jcol) zl += __expf(acc[r] - newm);
            zsum = zl;
            m = newm;
        }
    }
    // merge lane l with lane l+16 (same column, other row-half)
    const float mo = __shfl_xor(m, 16, 32);
    const float zo = __shfl_xor(zsum, 16, 32);
    const float M = fmaxf(m, mo);
    float Z = 0.0f;
    if (M != -__builtin_inff())
        Z = zsum * __expf(m - M) + zo * __expf(mo - M);
    if (lane < 16) {
        colmax[b * WINDOW + jcol] = M;
        colsum[b * WINDOW + jcol] = Z;
    }
}

// ---------------------------------------------------------------------------
// Kernel 3: out[b,i,:] = sum_{j<=i} P[i,j] * V[j,:].
// Dynamic LDS carve-up:
//   Vtile [1024][32] bf16 (64 KiB)  <- filled by TDM each j-step
//   Apart [8][256]   f32  (8 KiB)   <- per-wave partial score tiles
//   Ptile [16][32]   bf16 (1 KiB)   <- normalized probabilities
// Per 32-column j-step: sync; wave0 issues tensor_load_to_lds for the V
// slab (overlaps with the score phase); score phase; sync; normalize; wave0
// s_wait_tensorcnt 0; sync; PV phase reads P and V fragments from LDS.
// ---------------------------------------------------------------------------
#define VTILE_BYTES (EMBED * 32 * 2)              // 65536
#define APART_BYTES (8 * 256 * 4)                 // 8192
#define K3_LDS_BYTES (VTILE_BYTES + APART_BYTES + 16 * 32 * 2)

__global__ __launch_bounds__(256) void attn_out_kernel(
    const __bf16* __restrict__ Qb, const __bf16* __restrict__ Kb,
    const __bf16* __restrict__ Vt,
    const float* __restrict__ colmax, const float* __restrict__ colsum,
    float* __restrict__ out)
{
    extern __shared__ char smem[];
    __bf16* Vtile = (__bf16*)smem;                       // [1024][32]
    float*  Apart = (float*)(smem + VTILE_BYTES);        // [8][256]
    __bf16* Ptile = (__bf16*)(smem + VTILE_BYTES + APART_BYTES); // [16][32]

    const int tid  = threadIdx.x;
    const int lane = tid & 31;
    const int wave = tid >> 5;
    const int b  = blockIdx.x >> 7;
    const int it = blockIdx.x & 127;

    const __bf16* Q = Qb + (size_t)b * WINDOW * EMBED;
    const __bf16* K = Kb + (size_t)b * WINDOW * EMBED;
    const __bf16* V = Vt + (size_t)b * EMBED * WINDOW;   // [e][s]
    const float* cm = colmax + b * WINDOW;
    const float* cz = colsum + b * WINDOW;

    const unsigned vtile_lds = (unsigned)(uintptr_t)Vtile; // LDS byte offset

    const int nsub  = wave & 1;   // which 16-col j subtile (score phase)
    const int kc    = wave >> 1;  // which 256-wide E chunk  (score phase)
    const int ebase = wave * 128; // output column strip     (PV phase)

    v8f c[8];
#pragma unroll
    for (int s = 0; s < 8; ++s)
#pragma unroll
        for (int i = 0; i < 8; ++i) c[s][i] = 0.0f;

    const int itop = it * 16 + 15;
    for (int jb = 0; jb <= itop; jb += 32) {
        __syncthreads();   // previous PV reads of Vtile are done
        if (wave == 0) {
            // DMA V slab: 32 s-columns x 1024 e-lines -> Vtile[e][0..31]
            tdm_load_2d_bf16(vtile_lds, V + jb,
                             /*tdim0=*/WINDOW, /*tdim1=*/EMBED,
                             /*tile0=*/32, /*tile1=*/EMBED,
                             /*stride0=*/WINDOW);
        }

        // ---- score phase: partial QK^T for columns [jb, jb+32) ----
        v8f acc;
#pragma unroll
        for (int i = 0; i < 8; ++i) acc[i] = 0.0f;
        const int jcolbase = jb + nsub * 16;
#pragma unroll 2
        for (int kk = 0; kk < 8; ++kk) {
            const int ko = kc * 256 + kk * 32;
            v16bf a = loadA_bf16(Q, EMBED, it * 16, ko, lane);
            v16bf bb = loadB_bf16(K, EMBED, jcolbase, ko, lane);
            acc = WMMA_BF16(a, bb, acc);
        }
        {
            const int n  = lane & 15;
            const int mb = (lane >> 4) ? 8 : 0;
#pragma unroll
            for (int r = 0; r < 8; ++r)
                Apart[wave * 256 + (mb + r) * 16 + n] = acc[r];
        }
        __syncthreads();

        // ---- normalize: 512 elements, 2 per thread ----
#pragma unroll
        for (int e2 = 0; e2 < 2; ++e2) {
            const int elem = tid + e2 * 256;     // 0..511
            const int ns   = elem >> 8;          // j subtile
            const int idx  = elem & 255;
            const float v = Apart[(ns + 0) * 256 + idx] + Apart[(ns + 2) * 256 + idx] +
                            Apart[(ns + 4) * 256 + idx] + Apart[(ns + 6) * 256 + idx];
            const int mloc = idx >> 4, n = idx & 15;
            const int i = it * 16 + mloc;
            const int j = jb + ns * 16 + n;
            float p = 0.0f;
            if (j <= i) p = __expf(v - cm[j]) / cz[j];
            Ptile[mloc * 32 + ns * 16 + n] = (__bf16)p;
        }
        if (wave == 0) __builtin_amdgcn_s_wait_tensorcnt(0);
        __syncthreads();   // Ptile ready AND Vtile DMA complete

        // ---- PV phase: out-strip += P(16x32) x Vtile(32x128) ----
        v16bf pf = loadA_bf16(Ptile, 32, 0, 0, lane);
#pragma unroll
        for (int s = 0; s < 8; ++s) {
            v16bf vb = loadB_bf16(Vtile, 32, ebase + s * 16, 0, lane);
            c[s] = WMMA_BF16(pf, vb, c[s]);
        }
    }

    // ---- store f32 output ----
    const int n  = lane & 15;
    const int mb = it * 16 + ((lane >> 4) ? 8 : 0);
#pragma unroll
    for (int s = 0; s < 8; ++s) {
        const int ecol = ebase + s * 16 + n;
#pragma unroll
        for (int r = 0; r < 8; ++r)
            out[((size_t)b * WINDOW + mb + r) * EMBED + ecol] = c[s][r];
    }
}

// ---------------------------------------------------------------------------
extern "C" void kernel_launch(void* const* d_in, const int* in_sizes, int n_in,
                              void* d_out, int out_size, void* d_ws, size_t ws_size,
                              hipStream_t stream) {
    const float* X  = (const float*)d_in[0];
    const float* Wq = (const float*)d_in[1];
    const float* bq = (const float*)d_in[2];
    const float* Wk = (const float*)d_in[3];
    const float* bk = (const float*)d_in[4];
    const float* Wv = (const float*)d_in[5];
    const float* bv = (const float*)d_in[6];
    float* out = (float*)d_out;

    char* ws = (char*)d_ws;
    const size_t QKV_BYTES = (size_t)BATCH * WINDOW * EMBED * sizeof(__bf16); // 32 MiB
    __bf16* Qb = (__bf16*)(ws);
    __bf16* Kb = (__bf16*)(ws + QKV_BYTES);
    __bf16* Vt = (__bf16*)(ws + 2 * QKV_BYTES);
    float* colmax = (float*)(ws + 3 * QKV_BYTES);
    float* colsum = (float*)(ws + 3 * QKV_BYTES + (size_t)BATCH * WINDOW * sizeof(float));

    // 1) QKV projections (M=B*S=16384, N=E=1024): block tile 32x64, z = q/k/v
    dim3 g1(16384 / 32, 1024 / 64, 3);
    qkv_kernel<<<g1, 256, 0, stream>>>(X, Wq, bq, Wk, bk, Wv, bv, Qb, Kb, Vt);

    // 2) column softmax stats: 1024 waves = 128 blocks of 8
    stats_kernel<<<128, 256, 0, stream>>>(Qb, Kb, colmax, colsum);

    // 3) output: one block per (batch, 16-row tile), 73 KiB dynamic LDS
    attn_out_kernel<<<BATCH * (WINDOW / 16), 256, K3_LDS_BYTES, stream>>>(
        Qb, Kb, Vt, colmax, colsum, out);
}
